// AMM_70403103916208
// MI455X (gfx1250) — matrix operand order repaired and checked by reading, not compiled, not run on hardware
//
#include <hip/hip_runtime.h>
#include <hip/hip_bf16.h>
#include <math.h>

typedef __attribute__((ext_vector_type(16))) _Float16 v16h;
typedef __attribute__((ext_vector_type(8)))  _Float16 v8h;
typedef __attribute__((ext_vector_type(8)))  float    v8f;

#define HW     4096
#define KREAL  392
#define KP     416      // padded to 13*32
#define KCH    13       // number of K chunks of 32
#define CF     256      // visual feature channels
#define NWAVE  8        // waves per 256-thread block (wave32)
#define NTILE  (HW/16)  // 256 16-wide tiles

union V16U { v16h v; v8h h[2]; };

// A matrix 16x32 f16 fragment: lane m(=lane&15), hi=lane>>4.
// VGPR0..3 hold K = hi*8 + {0..7}; VGPR4..7 hold K = 16 + hi*8 + {0..7}.
__device__ __forceinline__ v16h load_a_frag(const _Float16* rowp, int kk, int hi) {
    V16U u;
    u.h[0] = *(const v8h*)(rowp + kk*32 + hi*8);
    u.h[1] = *(const v8h*)(rowp + kk*32 + 16 + hi*8);
    return u.v;
}
// B matrix 32x16 f16 fragment: lane n holds K = hi*8 + {0..7}, then K = 16 + hi*8 + {0..7}.
__device__ __forceinline__ v16h load_b_frag(const _Float16* colp, int kk, int hi) {
    V16U u;
    u.h[0] = *(const v8h*)(colp + kk*32 + hi*8);
    u.h[1] = *(const v8h*)(colp + kk*32 + 16 + hi*8);
    return u.v;
}

// ---- prep: fps packed [4096][416] f16 (raw-reshape rows of scaled concat) ----
__global__ void build_fps(const float* __restrict__ fs, const float* __restrict__ ls,
                          _Float16* __restrict__ fpsP) {
    int idx = blockIdx.x * 256 + threadIdx.x;     // [0, 4096*416)
    int i = idx / KP;
    int j = idx - i * KP;
    float v = 0.f;
    if (j < KREAL) {
        int f = i * KREAL + j;                    // flat index into concat buffer
        v = (f < CF * HW) ? 0.01f * fs[f] : ls[f - CF * HW];
    }
    fpsP[idx] = (_Float16)v;
}

// ---- prep: fpr transposed [4096(q)][416(k)] f16 (coalesced reads) ----
__global__ void build_fpr(const float* __restrict__ fr, const float* __restrict__ lr,
                          _Float16* __restrict__ fprT) {
    int idx = blockIdx.x * 256 + threadIdx.x;     // j-major: idx = j*4096 + q
    int j = idx >> 12;
    int q = idx & (HW - 1);
    float v = 0.f;
    if (j < KREAL) v = (j < CF) ? 0.01f * fr[idx] : lr[idx - CF * HW];
    fprT[(size_t)q * KP + j] = (_Float16)v;
}

// ---- prep: beta/gama 1x1 convs + column mask -> mb = M*beta, mg = M*gama ----
__global__ void build_vec(const float* __restrict__ fr,
                          const float* __restrict__ w1, const float* __restrict__ b1,
                          const float* __restrict__ w2, const float* __restrict__ b2,
                          const int* __restrict__ msk_s, const int* __restrict__ msk_r,
                          float* __restrict__ mb, float* __restrict__ mg) {
    int q = blockIdx.x * 256 + threadIdx.x;
    float a1 = b1[0], a2 = b2[0];
    for (int c = 0; c < CF; ++c) {
        float f = fr[c * HW + q];                 // coalesced across q
        a1 = fmaf(f, w1[c], a1);
        a2 = fmaf(f, w2[c], a2);
    }
    float m = (msk_r[q] == msk_s[q]) ? 1.0f : 0.0f;
    mb[q] = m * a1;
    mg[q] = m * a2;
}

// ---- pass A: per-column online softmax stats (m_q, Z_q) over all rows i ----
__global__ void __launch_bounds__(256)
col_stats(const _Float16* __restrict__ fpsP, const _Float16* __restrict__ fprT,
          const float* __restrict__ mb, const float* __restrict__ mg,
          float* __restrict__ mcol, float* __restrict__ wb, float* __restrict__ wg) {
    int tid  = threadIdx.x;
    int lane = tid & 31, wave = tid >> 5;
    int n = lane & 15, hi = lane >> 4;
    int q0 = blockIdx.x * 16;

    // B fragments for this 16-column tile stay resident across the whole i loop
    v16h bfrag[KCH];
    const _Float16* colp = fprT + (size_t)(q0 + n) * KP;
    #pragma unroll
    for (int kk = 0; kk < KCH; ++kk) bfrag[kk] = load_b_frag(colp, kk, hi);

    float mrun = -3.0e38f, zrun = 0.f;
    for (int t = wave; t < NTILE; t += NWAVE) {
        const _Float16* rowp = fpsP + (size_t)(t * 16 + n) * KP;
        v8f c = {};
        #pragma unroll
        for (int kk = 0; kk < KCH; ++kk) {
            v16h a = load_a_frag(rowp, kk, hi);
            c = __builtin_amdgcn_wmma_f32_16x16x32_f16(false, a, false, bfrag[kk],
                                                       (short)0, c, false, false);
        }
        // lane holds 8 rows of one column; lane^16 holds the other 8 rows
        float tmax = c[0];
        #pragma unroll
        for (int r = 1; r < 8; ++r) tmax = fmaxf(tmax, c[r]);
        tmax = fmaxf(tmax, __shfl_xor(tmax, 16, 32));
        float nm = fmaxf(mrun, tmax);
        float ts = 0.f;
        #pragma unroll
        for (int r = 0; r < 8; ++r) ts += __expf(c[r] - nm);
        ts += __shfl_xor(ts, 16, 32);
        zrun = zrun * __expf(mrun - nm) + ts;
        mrun = nm;
    }
    __shared__ float sm[NWAVE][16], sz[NWAVE][16];
    if (hi == 0) { sm[wave][n] = mrun; sz[wave][n] = zrun; }
    __syncthreads();
    if (tid < 16) {  // deterministic fixed-order cross-wave combine
        float m = sm[0][tid];
        for (int w = 1; w < NWAVE; ++w) m = fmaxf(m, sm[w][tid]);
        float z = 0.f;
        for (int w = 0; w < NWAVE; ++w) z += sz[w][tid] * __expf(sm[w][tid] - m);
        int q = q0 + tid;
        mcol[q] = m;
        float inv = 1.0f / z;
        wb[q] = mb[q] * inv;   // M[q]*beta[q]/Z_q
        wg[q] = mg[q] * inv;
    }
}

// ---- pass B: beta_hat[i] = sum_q exp(S[i,q]-m_q) * wb[q]  (and gama_hat) ----
__global__ void __launch_bounds__(256)
row_pass(const _Float16* __restrict__ fpsP, const _Float16* __restrict__ fprT,
         const float* __restrict__ mcol, const float* __restrict__ wb,
         const float* __restrict__ wg,
         float* __restrict__ bhat, float* __restrict__ ghat) {
    int tid  = threadIdx.x;
    int lane = tid & 31, wave = tid >> 5;
    int n = lane & 15, hi = lane >> 4;
    int i0 = blockIdx.x * 16;

    // A fragments for this 16-row tile stay resident across the whole q loop
    v16h afrag[KCH];
    const _Float16* rowp = fpsP + (size_t)(i0 + n) * KP;
    #pragma unroll
    for (int kk = 0; kk < KCH; ++kk) afrag[kk] = load_a_frag(rowp, kk, hi);

    float accb[8] = {}, accg[8] = {};
    for (int t = wave; t < NTILE; t += NWAVE) {
        int q = t * 16 + n;
        float mq = mcol[q], wbq = wb[q], wgq = wg[q];
        const _Float16* colp = fprT + (size_t)q * KP;
        v8f c = {};
        #pragma unroll
        for (int kk = 0; kk < KCH; ++kk) {
            v16h b = load_b_frag(colp, kk, hi);
            c = __builtin_amdgcn_wmma_f32_16x16x32_f16(false, afrag[kk], false, b,
                                                       (short)0, c, false, false);
        }
        #pragma unroll
        for (int r = 0; r < 8; ++r) {
            float e = __expf(c[r] - mq);
            accb[r] = fmaf(e, wbq, accb[r]);
            accg[r] = fmaf(e, wgq, accg[r]);
        }
    }
    // deterministic reduction: butterfly within each 16-lane half (same rows)
    #pragma unroll
    for (int r = 0; r < 8; ++r) {
        #pragma unroll
        for (int m = 8; m >= 1; m >>= 1) {
            accb[r] += __shfl_xor(accb[r], m, 16);
            accg[r] += __shfl_xor(accg[r], m, 16);
        }
    }
    __shared__ float pb[NWAVE * 2][8], pg[NWAVE * 2][8];
    int half = wave * 2 + hi;
    if (n == 0) {
        #pragma unroll
        for (int r = 0; r < 8; ++r) { pb[half][r] = accb[r]; pg[half][r] = accg[r]; }
    }
    __syncthreads();
    if (tid < 16) {  // row tid = r + 8*rhi ; fixed-order sum over waves
        int rhi = tid >> 3, r = tid & 7;
        float sb = 0.f, sg = 0.f;
        for (int w = 0; w < NWAVE; ++w) {
            sb += pb[w * 2 + rhi][r];
            sg += pg[w * 2 + rhi][r];
        }
        bhat[i0 + tid] = sb;
        ghat[i0 + tid] = sg;
    }
}

// ---- epilogue: out[c,p] = gama_hat[p]*feat_src[c,p] + beta_hat[p] ----
__global__ void finalize(const float* __restrict__ fs, const float* __restrict__ bhat,
                         const float* __restrict__ ghat, float* __restrict__ out) {
    int idx = blockIdx.x * 256 + threadIdx.x;     // [0, 256*4096)
    int p = idx & (HW - 1);
    out[idx] = fmaf(ghat[p], fs[idx], bhat[p]);
}

extern "C" void kernel_launch(void* const* d_in, const int* in_sizes, int n_in,
                              void* d_out, int out_size, void* d_ws, size_t ws_size,
                              hipStream_t stream) {
    const float* feat_src = (const float*)d_in[0];
    const float* feat_ref = (const float*)d_in[1];
    const float* lmk_src  = (const float*)d_in[2];
    const float* lmk_ref  = (const float*)d_in[3];
    const int*   mask_src = (const int*)d_in[4];
    const int*   mask_ref = (const int*)d_in[5];
    const float* w1 = (const float*)d_in[6];
    const float* b1 = (const float*)d_in[7];
    const float* w2 = (const float*)d_in[8];
    const float* b2 = (const float*)d_in[9];
    float* out = (float*)d_out;

    char* ws = (char*)d_ws;
    const size_t PACK = (size_t)HW * KP * sizeof(_Float16);   // 3,407,872 B (256B aligned)
    _Float16* fpsP = (_Float16*)ws;
    _Float16* fprT = (_Float16*)(ws + PACK);
    float* mb   = (float*)(ws + 2 * PACK);
    float* mg   = mb   + HW;
    float* mcol = mg   + HW;
    float* wb   = mcol + HW;
    float* wg   = wb   + HW;
    float* bhat = wg   + HW;
    float* ghat = bhat + HW;

    const int PACK_BLOCKS = (HW * KP) / 256;   // 6656
    build_fps<<<PACK_BLOCKS, 256, 0, stream>>>(feat_src, lmk_src, fpsP);
    build_fpr<<<PACK_BLOCKS, 256, 0, stream>>>(feat_ref, lmk_ref, fprT);
    build_vec<<<HW / 256, 256, 0, stream>>>(feat_ref, w1, b1, w2, b2,
                                            mask_src, mask_ref, mb, mg);
    col_stats<<<HW / 16, 256, 0, stream>>>(fpsP, fprT, mb, mg, mcol, wb, wg);
    row_pass<<<HW / 16, 256, 0, stream>>>(fpsP, fprT, mcol, wb, wg, bhat, ghat);
    finalize<<<(CF * HW) / 256, 256, 0, stream>>>(feat_src, bhat, ghat, out);
}
